// MultiHeadAttention_1288490188995
// MI455X (gfx1250) — matrix operand order repaired and checked by reading, not compiled, not run on hardware
//
#include <hip/hip_runtime.h>

// ---------------- problem constants ----------------
// D_MODEL=768, D_K=D_V=128, N_HEAD=6, B=4, S=2048
#define BB    4
#define SS    2048
#define DM    768
#define NH    6
#define DK    128
#define NEGV  (-1.0e9f)
#define SCALE 0.08838834764831845f   // 1/sqrt(128)

typedef __attribute__((ext_vector_type(16))) __bf16 v16bf;
typedef __attribute__((ext_vector_type(8)))  __bf16 v8bf;
typedef __attribute__((ext_vector_type(8)))  float  v8f;

// ---------------- WMMA fragment loaders (from LDS tiles) ----------------
// A operand: 16x32 bf16, source row-major [row][k], row stride `ld` halves.
// lane l (l<16): row=l, K = 0..7 and 16..23 ; lane l+16: row=l, K = 8..15 and 24..31
__device__ inline v16bf load_frag_a(const __bf16* base, int ld, int row0, int k0) {
  int lane = threadIdx.x & 31;
  int r  = row0 + (lane & 15);
  int kh = lane >> 4;
  const __bf16* p = base + r * ld + k0 + kh * 8;
  v8bf lo = *(const v8bf*)p;
  v8bf hi = *(const v8bf*)(p + 16);
  return __builtin_shufflevector(lo, hi, 0,1,2,3,4,5,6,7,8,9,10,11,12,13,14,15);
}

// B operand: 32x16 bf16, source column-major, i.e. tile stored [n][k], stride `ld`.
// lane l (l<16): col=l, K = 0..7 and 16..23 ; lane l+16: col=l, K = 8..15 and 24..31
__device__ inline v16bf load_frag_b(const __bf16* base, int ld, int n0, int k0) {
  int lane = threadIdx.x & 31;
  int n  = n0 + (lane & 15);
  int kh = lane >> 4;
  const __bf16* p = base + n * ld + k0 + kh * 8;
  v8bf lo = *(const v8bf*)p;
  v8bf hi = *(const v8bf*)(p + 16);
  return __builtin_shufflevector(lo, hi, 0,1,2,3,4,5,6,7,8,9,10,11,12,13,14,15);
}

__device__ inline v8f wmma_bf16(v16bf a, v16bf b, v8f c) {
  return __builtin_amdgcn_wmma_f32_16x16x32_bf16(false, a, false, b, (short)0, c,
                                                 false, false);
}

// ---------------- kernel 1: weight transpose + f32->bf16 ----------------
// src: [K=768][N=768] row-major f32 ; dst: [N=768][K=768] bf16 (B^T layout)
__global__ __launch_bounds__(256) void k_wt(const float* __restrict__ src,
                                            __bf16* __restrict__ dst) {
  int i = blockIdx.x * 256 + threadIdx.x;        // grid exactly covers 768*768
  int n = i / DM, k = i % DM;
  dst[(size_t)n * DM + k] = (__bf16)src[(size_t)k * DM + n];
}

// ---------------- kernel 2/4: GEMM  C[M=8192][N=768] = A[M][768] * B^T + ...
// Double-buffered LDS + register prefetch: one barrier per 32-wide K step.
// mode 0: q proj  -> bf16 [b][h][s][dk]
// mode 1: k proj  -> bf16 [b][h][s][dk]
// mode 2: v proj  -> bf16 [b][h][dk][s]   (pre-transposed for attn B-operand)
// mode 3: out proj-> f32  acc + bias + resid  (A is bf16 ctx)
__global__ __launch_bounds__(256) void k_gemm(const float* __restrict__ A32,
                                              const __bf16* __restrict__ A16,
                                              const __bf16* __restrict__ BT,
                                              const float* __restrict__ bias,
                                              const float* __restrict__ resid,
                                              __bf16* __restrict__ dst_bf,
                                              float* __restrict__ dst_f,
                                              int mode) {
  __shared__ __align__(32) __bf16 As[2][128 * 48];   // 2 x (128x32, pad 48)
  __shared__ __align__(32) __bf16 Bs[2][64 * 48];    // 2 x (64x32,  pad 48)
  int tid = threadIdx.x;
  int lane = tid & 31, wid = tid >> 5;
  int wm = wid & 3, wn = wid >> 2;                   // 4 x 2 wave grid (32x32 each)
  int m0 = blockIdx.x * 128, n0 = blockIdx.y * 64;

  // per-thread staging coordinates (fixed for all K steps)
  const int ar = tid >> 1, ac = (tid & 1) * 16;      // A: 2 thr/row, 16 elems
  const int bn = tid >> 2, bc = (tid & 3) * 8;       // B: 4 thr/row, 8 halves

  // ---- prologue: fetch K-tile 0 into registers ----
  float fa[16];                                      // A staging regs (mode<3)
  v8bf  a16lo, a16hi;                                // A staging regs (mode 3)
  v8bf  brg;                                         // B staging regs
  if (mode < 3) {
    const float* s = A32 + (size_t)(m0 + ar) * DM + ac;
    #pragma unroll
    for (int i = 0; i < 16; ++i) fa[i] = s[i];
  } else {
    const __bf16* s = A16 + (size_t)(m0 + ar) * DM + ac;
    a16lo = *(const v8bf*)s;
    a16hi = *(const v8bf*)(s + 8);
  }
  brg = *(const v8bf*)(BT + (size_t)(n0 + bn) * DM + bc);

  v8f acc[2][2] = {};
  int buf = 0;
  for (int kt = 0; kt < DM / 32; ++kt) {
    // ---- commit staged registers into LDS buffer `buf` ----
    if (mode < 3) {
      #pragma unroll
      for (int i = 0; i < 16; ++i) As[buf][ar * 48 + ac + i] = (__bf16)fa[i];
    } else {
      *(v8bf*)&As[buf][ar * 48 + ac]     = a16lo;
      *(v8bf*)&As[buf][ar * 48 + ac + 8] = a16hi;
    }
    *(v8bf*)&Bs[buf][bn * 48 + bc] = brg;
    __syncthreads();

    // ---- prefetch next K-tile into registers (overlaps with WMMA) ----
    if (kt + 1 < DM / 32) {
      int k0 = (kt + 1) * 32;
      if (mode < 3) {
        const float* s = A32 + (size_t)(m0 + ar) * DM + k0 + ac;
        #pragma unroll
        for (int i = 0; i < 16; ++i) fa[i] = s[i];
      } else {
        const __bf16* s = A16 + (size_t)(m0 + ar) * DM + k0 + ac;
        a16lo = *(const v8bf*)s;
        a16hi = *(const v8bf*)(s + 8);
      }
      brg = *(const v8bf*)(BT + (size_t)(n0 + bn) * DM + k0 + bc);
    }

    // ---- WMMA on current buffer ----
    #pragma unroll
    for (int i = 0; i < 2; ++i) {
      v16bf a = load_frag_a(As[buf], 48, wm * 32 + i * 16, 0);
      #pragma unroll
      for (int j = 0; j < 2; ++j) {
        v16bf b = load_frag_b(Bs[buf], 48, wn * 32 + j * 16, 0);
        acc[i][j] = wmma_bf16(a, b, acc[i][j]);
      }
    }
    buf ^= 1;
  }
  // epilogue: D-frag element (lane,e): row = e + 8*(lane>>4), col = lane&15
  #pragma unroll
  for (int i = 0; i < 2; ++i)
    #pragma unroll
    for (int j = 0; j < 2; ++j)
      #pragma unroll
      for (int e = 0; e < 8; ++e) {
        int rl = wm * 32 + i * 16 + e + ((lane >> 4) << 3);
        int cl = wn * 32 + j * 16 + (lane & 15);
        int m = m0 + rl, n = n0 + cl;
        float v = acc[i][j][e] + bias[n];
        if (mode == 3) {
          dst_f[(size_t)m * DM + n] = v + resid[(size_t)m * DM + n];
        } else {
          int b = m >> 11, s = m & (SS - 1);
          int h = n >> 7, dk = n & (DK - 1);
          if (mode == 2)
            dst_bf[(((size_t)b * NH + h) * DK + dk) * SS + s] = (__bf16)v;   // V^T
          else
            dst_bf[(((size_t)b * NH + h) * SS + s) * DK + dk] = (__bf16)v;   // Q/K
        }
      }
}

// ---------------- kernel 3: attention per (b,h, 64-row q block) ----------------
__global__ __launch_bounds__(256) void k_attn(const __bf16* __restrict__ qg,
                                              const __bf16* __restrict__ kg,
                                              const __bf16* __restrict__ vtg,
                                              const unsigned char* __restrict__ mask,
                                              float* __restrict__ attn,
                                              __bf16* __restrict__ ctx) {
  __shared__ __align__(32) __bf16 qs[64 * 136];   // q tile 64x128  (17.0 KB)
  __shared__ __align__(32) __bf16 kv[128 * 72];   // k tile 64x128 / vT tile 128x64
  __shared__ float sc[64 * 66];                   // score tile 64x64 f32 (16.5 KB)
  __shared__ __align__(32) __bf16 pt[64 * 72];    // prob tile bf16 (9 KB)
  __shared__ float rowm[64], rowl[64];

  int tid = threadIdx.x, lane = tid & 31, wid = tid >> 5;
  int wm = wid & 3, wn = wid >> 2;
  int bh = blockIdx.y;                 // 0..23
  int b = bh / NH, h = bh % NH;
  int q0 = blockIdx.x * 64;
  size_t hb = (size_t)bh;

  {   // stage q tile
    int r = tid >> 2, c0 = (tid & 3) * 32;
    const __bf16* s = qg + (hb * SS + q0 + r) * DK + c0;
    #pragma unroll
    for (int i = 0; i < 4; ++i)
      *(v8bf*)&qs[r * 136 + c0 + i * 8] = *(const v8bf*)(s + i * 8);
  }
  if (tid < 64) { rowm[tid] = -3.0e38f; rowl[tid] = 0.f; }
  __syncthreads();

  // ---- pass 1: scores, online max/sum, spill raw scores to attn buffer ----
  for (int kb = 0; kb < SS / 64; ++kb) {
    int k0 = kb * 64;
    {   // stage K tile row-major [krow][d]  == B-operand column-major
      int r = tid >> 2, c0 = (tid & 3) * 32;
      const __bf16* s = kg + (hb * SS + k0 + r) * DK + c0;
      #pragma unroll
      for (int i = 0; i < 4; ++i)
        *(v8bf*)&kv[r * 136 + c0 + i * 8] = *(const v8bf*)(s + i * 8);
      // prefetch next K tile (lands in L2/WGP$, hidden under WMMA)
      if (kb + 1 < SS / 64)
        __builtin_prefetch(kg + (hb * SS + k0 + 64 + r) * DK + c0, 0, 0);
    }
    __syncthreads();

    v8f acc[2] = {};
    #pragma unroll
    for (int ks = 0; ks < 4; ++ks) {     // d = 128 = 4 x 32
      v16bf a = load_frag_a(qs, 136, wm * 16, ks * 32);
      #pragma unroll
      for (int j = 0; j < 2; ++j) {
        v16bf bf = load_frag_b(kv, 136, wn * 32 + j * 16, ks * 32);
        acc[j] = wmma_bf16(a, bf, acc[j]);
      }
    }
    // scale + mask, drop into LDS score tile
    #pragma unroll
    for (int j = 0; j < 2; ++j)
      #pragma unroll
      for (int e = 0; e < 8; ++e) {
        int rl = wm * 16 + e + ((lane >> 4) << 3);
        int cl = wn * 32 + j * 16 + (lane & 15);
        float v = acc[j][e] * SCALE;
        if (mask[((size_t)b * SS + q0 + rl) * SS + k0 + cl]) v = NEGV;
        sc[rl * 66 + cl] = v;
      }
    __syncthreads();

    if (tid < 64) {   // online softmax stats per row
      int r = tid;
      float mx = -3.0e38f;
      for (int c = 0; c < 64; ++c) mx = fmaxf(mx, sc[r * 66 + c]);
      float mo = rowm[r], mn = fmaxf(mo, mx);
      float s = 0.f;
      for (int c = 0; c < 64; ++c) s += __expf(sc[r * 66 + c] - mn);
      rowl[r] = rowl[r] * __expf(mo - mn) + s;
      rowm[r] = mn;
    }
    {   // spill raw scores (coalesced)
      int r = tid >> 2, c0 = (tid & 3) * 16;
      float* d = attn + (hb * SS + q0 + r) * SS + k0 + c0;
      #pragma unroll
      for (int i = 0; i < 16; ++i) d[i] = sc[r * 66 + c0 + i];
    }
    __syncthreads();
  }

  // ---- pass 2: normalize -> attn, and ctx = P @ V via WMMA ----
  v8f acc2[4] = {};
  for (int kb = 0; kb < SS / 64; ++kb) {
    int k0 = kb * 64;
    {   // stage V^T tile [d][s]  == B-operand column-major
      int dc = tid >> 1, c0 = (tid & 1) * 32;
      const __bf16* s = vtg + (hb * DK + dc) * SS + k0 + c0;
      #pragma unroll
      for (int i = 0; i < 4; ++i)
        *(v8bf*)&kv[dc * 72 + c0 + i * 8] = *(const v8bf*)(s + i * 8);
      if (kb + 1 < SS / 64)
        __builtin_prefetch(vtg + (hb * DK + dc) * SS + k0 + 64 + c0, 0, 0);
    }
    {   // read raw scores back, write probabilities, stage bf16 P tile
      int r = tid >> 2, c0 = (tid & 3) * 16;
      float* ap = attn + (hb * SS + q0 + r) * SS + k0 + c0;
      float m = rowm[r], l = rowl[r];
      float inv = l > 0.f ? 1.f / l : 0.f;
      #pragma unroll
      for (int i = 0; i < 16; ++i) {
        float p = __expf(ap[i] - m) * inv;
        ap[i] = p;
        pt[r * 72 + c0 + i] = (__bf16)p;
      }
      if (kb + 1 < SS / 64)
        __builtin_prefetch(ap + 64, 0, 0);
    }
    __syncthreads();
    #pragma unroll
    for (int ks = 0; ks < 2; ++ks) {    // 64-wide k block = 2 x 32
      v16bf a = load_frag_a(pt, 72, wm * 16, ks * 32);
      #pragma unroll
      for (int j = 0; j < 4; ++j) {
        v16bf bf = load_frag_b(kv, 72, wn * 64 + j * 16, ks * 32);
        acc2[j] = wmma_bf16(a, bf, acc2[j]);
      }
    }
    __syncthreads();
  }
  // store ctx as bf16 [b][s][h*128+d]
  #pragma unroll
  for (int j = 0; j < 4; ++j)
    #pragma unroll
    for (int e = 0; e < 8; ++e) {
      int rl = wm * 16 + e + ((lane >> 4) << 3);
      int dc = wn * 64 + j * 16 + (lane & 15);
      ctx[((size_t)b * SS + q0 + rl) * DM + h * DK + dc] = (__bf16)acc2[j][e];
    }
}

// ---------------- kernel 5: LayerNorm per row ----------------
__global__ __launch_bounds__(256) void k_ln(const float* __restrict__ x,
                                            const float* __restrict__ g,
                                            const float* __restrict__ bta,
                                            float* __restrict__ out) {
  __shared__ float s1[256], s2[256];
  int row = blockIdx.x, tid = threadIdx.x;
  const float* xr = x + (size_t)row * DM;
  float a = 0.f, q = 0.f;
  #pragma unroll
  for (int i = 0; i < 3; ++i) { float v = xr[tid + i * 256]; a += v; q += v * v; }
  s1[tid] = a; s2[tid] = q;
  __syncthreads();
  for (int st = 128; st > 0; st >>= 1) {
    if (tid < st) { s1[tid] += s1[tid + st]; s2[tid] += s2[tid + st]; }
    __syncthreads();
  }
  float mu = s1[0] * (1.f / DM);
  float var = s2[0] * (1.f / DM) - mu * mu;
  float rs = rsqrtf(var + 1e-5f);
  #pragma unroll
  for (int i = 0; i < 3; ++i) {
    int c = tid + i * 256;
    out[(size_t)row * DM + c] = (xr[c] - mu) * rs * g[c] + bta[c];
  }
}

// ---------------- host launcher ----------------
extern "C" void kernel_launch(void* const* d_in, const int* in_sizes, int n_in,
                              void* d_out, int out_size, void* d_ws, size_t ws_size,
                              hipStream_t stream) {
  (void)in_sizes; (void)n_in; (void)out_size; (void)ws_size;
  const float* Q   = (const float*)d_in[0];
  const float* Kin = (const float*)d_in[1];
  const float* Vin = (const float*)d_in[2];
  const unsigned char* mask = (const unsigned char*)d_in[3];
  const float* Wq = (const float*)d_in[4];  const float* bq = (const float*)d_in[5];
  const float* Wk = (const float*)d_in[6];  const float* bk = (const float*)d_in[7];
  const float* Wv = (const float*)d_in[8];  const float* bv = (const float*)d_in[9];
  const float* Wo = (const float*)d_in[10]; const float* bo = (const float*)d_in[11];
  const float* lng = (const float*)d_in[12];
  const float* lnb = (const float*)d_in[13];

  char* w = (char*)d_ws;
  size_t o = 0;
  auto carve = [&](size_t bytes) {
    char* p = w + o;
    o += (bytes + 255) & ~(size_t)255;
    return p;
  };
  const size_t WTB  = (size_t)DM * DM * 2;                  // bf16 weight^T
  const size_t QKVB = (size_t)BB * NH * SS * DK * 2;        // bf16 q/k/vT
  __bf16* WqT = (__bf16*)carve(WTB);
  __bf16* WkT = (__bf16*)carve(WTB);
  __bf16* WvT = (__bf16*)carve(WTB);
  __bf16* WoT = (__bf16*)carve(WTB);
  __bf16* qws = (__bf16*)carve(QKVB);
  __bf16* kws = (__bf16*)carve(QKVB);
  __bf16* vtw = (__bf16*)carve(QKVB);
  __bf16* ctx = (__bf16*)carve((size_t)BB * SS * DM * 2);
  float*  pre = (float*)carve((size_t)BB * SS * DM * 4);

  float* outp  = (float*)d_out;
  float* attnp = outp + (size_t)BB * SS * DM;               // [B,H,S,S]

  const int wtBlocks = (DM * DM) / 256;                     // 2304
  k_wt<<<wtBlocks, 256, 0, stream>>>(Wq, WqT);
  k_wt<<<wtBlocks, 256, 0, stream>>>(Wk, WkT);
  k_wt<<<wtBlocks, 256, 0, stream>>>(Wv, WvT);
  k_wt<<<wtBlocks, 256, 0, stream>>>(Wo, WoT);

  dim3 gg((BB * SS) / 128, DM / 64);                        // (64, 12)
  k_gemm<<<gg, 256, 0, stream>>>(Q,   nullptr, WqT, bq, nullptr, qws, nullptr, 0);
  k_gemm<<<gg, 256, 0, stream>>>(Kin, nullptr, WkT, bk, nullptr, kws, nullptr, 1);
  k_gemm<<<gg, 256, 0, stream>>>(Vin, nullptr, WvT, bv, nullptr, vtw, nullptr, 2);

  k_attn<<<dim3(SS / 64, BB * NH), 256, 0, stream>>>(qws, kws, vtw, mask, attnp, ctx);

  k_gemm<<<gg, 256, 0, stream>>>(nullptr, ctx, WoT, bo, Q, nullptr, pre, 3);
  k_ln<<<BB * SS, 256, 0, stream>>>(pre, lng, lnb, outp);
}
